// ROGPL_79517024518975
// MI455X (gfx1250) — matrix operand order repaired and checked by reading, not compiled, not run on hardware
//
#include <hip/hip_runtime.h>

typedef __attribute__((ext_vector_type(16))) __bf16 v16bf;
typedef __attribute__((ext_vector_type(8)))  __bf16 v8bf;
typedef __attribute__((ext_vector_type(8)))  float  v8f;

// ---------------------------------------------------------------- utilities
__global__ void k_zero_f32(float* __restrict__ p, size_t n) {
  size_t i = (size_t)blockIdx.x * blockDim.x + threadIdx.x;
  size_t st = (size_t)gridDim.x * blockDim.x;
  for (; i < n; i += st) p[i] = 0.0f;
}

__global__ void k_zero_u32(unsigned* __restrict__ p, size_t n) {
  size_t i = (size_t)blockIdx.x * blockDim.x + threadIdx.x;
  size_t st = (size_t)gridDim.x * blockDim.x;
  for (; i < n; i += st) p[i] = 0u;
}

__global__ void k_f32_to_bf16(const float* __restrict__ src,
                              __bf16* __restrict__ dst, size_t n) {
  size_t i = (size_t)blockIdx.x * blockDim.x + threadIdx.x;
  size_t st = (size_t)gridDim.x * blockDim.x;
  for (; i < n; i += st) dst[i] = (__bf16)src[i];
}

// W[K][Nout] fp32 -> Wt[Nout][K] bf16 (so B fragments are contiguous K runs)
__global__ void k_tcvt(const float* __restrict__ W, __bf16* __restrict__ Wt,
                       int K, int Nout) {
  int i = blockIdx.x * blockDim.x + threadIdx.x;
  if (i >= K * Nout) return;
  int k = i / Nout, n = i % Nout;
  Wt[(size_t)n * K + k] = (__bf16)W[i];
}

// degree over dst (self-loop added later as +1)
__global__ void k_deg(const long long* __restrict__ ei, int Ee,
                      unsigned* __restrict__ deg) {
  int e = blockIdx.x * blockDim.x + threadIdx.x;
  if (e < Ee) atomicAdd(&deg[ei[(size_t)Ee + e]], 1u);
}

__global__ void k_dinv(const unsigned* __restrict__ deg,
                       float* __restrict__ dinv, int n) {
  int i = blockIdx.x * blockDim.x + threadIdx.x;
  if (i < n) dinv[i] = rsqrtf((float)deg[i] + 1.0f);
}

// ---------------------------------------------------- bf16 WMMA GEMM (core)
// C[M,Nc] bf16 = A[M,K] bf16 (row major)  x  Bt[Nc,K] bf16 (row major, = B^T)
// one wave -> 16x64 tile; block = 4 waves along N; fp32 accumulation.
__global__ __launch_bounds__(128) void k_gemm_bf16(
    const __bf16* __restrict__ A, const __bf16* __restrict__ Bt,
    __bf16* __restrict__ C, int M, int Nc, int K) {
  const int lane = threadIdx.x & 31;
  const int wave = threadIdx.x >> 5;
  const int m0 = blockIdx.y * 16;
  const int n0 = (blockIdx.x * 4 + wave) * 64;
  if (m0 >= M || n0 >= Nc) return;
  const int half = lane >> 4;   // 0: K-low group, 1: K-high group
  const int l16  = lane & 15;

  v8f acc0 = {}, acc1 = {}, acc2 = {}, acc3 = {};

  const __bf16* arow  = A  + (size_t)(m0 + l16) * K;
  const __bf16* brow0 = Bt + (size_t)(n0 + l16) * K + half * 16;
  const size_t bstep = (size_t)16 * K;   // next 16-column group of B

  for (int k0 = 0; k0 < K; k0 += 32) {
    // A fragment: elements 0..15 = K(k0+half*16..), contiguous
    v8bf alo = *(const v8bf*)(arow + k0 + half * 16);
    v8bf ahi = *(const v8bf*)(arow + k0 + half * 16 + 8);
    v16bf af = __builtin_shufflevector(alo, ahi,
        0, 1, 2, 3, 4, 5, 6, 7, 8, 9, 10, 11, 12, 13, 14, 15);
    // B fragments: lane holds 16 contiguous K values of one column
    v16bf b0 = *(const v16bf*)(brow0 + k0);
    v16bf b1 = *(const v16bf*)(brow0 + bstep + k0);
    v16bf b2 = *(const v16bf*)(brow0 + 2 * bstep + k0);
    v16bf b3 = *(const v16bf*)(brow0 + 3 * bstep + k0);
    acc0 = __builtin_amdgcn_wmma_f32_16x16x32_bf16(false, af, false, b0,
                                                   (short)0, acc0, false, false);
    acc1 = __builtin_amdgcn_wmma_f32_16x16x32_bf16(false, af, false, b1,
                                                   (short)0, acc1, false, false);
    acc2 = __builtin_amdgcn_wmma_f32_16x16x32_bf16(false, af, false, b2,
                                                   (short)0, acc2, false, false);
    acc3 = __builtin_amdgcn_wmma_f32_16x16x32_bf16(false, af, false, b3,
                                                   (short)0, acc3, false, false);
  }

  // C layout: VGPR r -> row m0 + half*8 + r, col n0 + l16 (+16/32/48)
  const int crow = m0 + half * 8;
#pragma unroll
  for (int r = 0; r < 8; ++r) {
    size_t ro = (size_t)(crow + r) * Nc + n0 + l16;
    C[ro]      = (__bf16)acc0[r];
    C[ro + 16] = (__bf16)acc1[r];
    C[ro + 32] = (__bf16)acc2[r];
    C[ro + 48] = (__bf16)acc3[r];
  }
}

// --------------------------------------------- edge gather + atomic scatter
// one wave per edge; lane covers D/32 contiguous features.
template <int D, bool MEAN>
__global__ __launch_bounds__(256) void k_edge(
    const long long* __restrict__ ei, int Ee,
    const __bf16* __restrict__ H, float* __restrict__ acc,
    const float* __restrict__ dinv) {
  int e = blockIdx.x * 8 + (threadIdx.x >> 5);
  if (e >= Ee) return;
  int lane = threadIdx.x & 31;
  long long s = ei[e];
  long long d = ei[(size_t)Ee + e];
  float norm = MEAN ? 1.0f : dinv[s] * dinv[d];
  constexpr int PL = D / 32;  // 16 or 8 features per lane
  const __bf16* hp = H + (size_t)s * D + lane * PL;
  float* ap = acc + (size_t)d * D + lane * PL;
#pragma unroll
  for (int c = 0; c < PL / 8; ++c) {
    v8bf v = *(const v8bf*)(hp + c * 8);
#pragma unroll
    for (int j = 0; j < 8; ++j)
      unsafeAtomicAdd(ap + c * 8 + j, (float)v[j] * norm);
  }
}

// ----------------------------------------------------------- finalize steps
// layer1: relu(agg + h*dinv^2 + b1) -> bf16 (input of layer 2)
__global__ void k_fin1(const float* __restrict__ agg, const __bf16* __restrict__ Hb,
                       const float* __restrict__ dinv, const float* __restrict__ b1,
                       __bf16* __restrict__ h1b, size_t total) {
  size_t i = (size_t)blockIdx.x * blockDim.x + threadIdx.x;
  size_t st = (size_t)gridDim.x * blockDim.x;
  for (; i < total; i += st) {
    int node = (int)(i >> 9), d = (int)(i & 511);
    float di = dinv[node];
    float v = agg[i] + (float)Hb[i] * di * di + b1[d];
    h1b[i] = (__bf16)fmaxf(v, 0.0f);
  }
}

// layer2: h2 = agg + h*dinv^2 + b2 -> xc[:, :256] (fp32 out) + bf16 copies
__global__ void k_fin2(const float* __restrict__ agg, const __bf16* __restrict__ Hb,
                       const float* __restrict__ dinv, const float* __restrict__ b2,
                       float* __restrict__ xc, __bf16* __restrict__ h2b,
                       __bf16* __restrict__ xcb, size_t total) {
  size_t i = (size_t)blockIdx.x * blockDim.x + threadIdx.x;
  size_t st = (size_t)gridDim.x * blockDim.x;
  for (; i < total; i += st) {
    int node = (int)(i >> 8), d = (int)(i & 255);
    float di = dinv[node];
    float v = agg[i] + (float)Hb[i] * di * di + b2[d];
    xc[(size_t)node * 512 + d] = v;
    h2b[i] = (__bf16)v;
    xcb[(size_t)node * 512 + d] = (__bf16)v;
  }
}

// neighbor mean -> xc[:, 256:512] + bf16 copy
__global__ void k_fin3(const float* __restrict__ nsum, const unsigned* __restrict__ deg,
                       float* __restrict__ xc, __bf16* __restrict__ xcb, size_t total) {
  size_t i = (size_t)blockIdx.x * blockDim.x + threadIdx.x;
  size_t st = (size_t)gridDim.x * blockDim.x;
  for (; i < total; i += st) {
    int node = (int)(i >> 8), d = (int)(i & 255);
    float v = nsum[i] / fmaxf((float)deg[node], 1.0f);
    xc[(size_t)node * 512 + 256 + d] = v;
    xcb[(size_t)node * 512 + 256 + d] = (__bf16)v;
  }
}

// ------------------------------------------- router + selected expert (wave/node)
__global__ __launch_bounds__(256) void k_expert(
    const __bf16* __restrict__ xcb, const float* __restrict__ Wp,
    const float* __restrict__ We, float* __restrict__ out, int Nn) {
  int node = blockIdx.x * 8 + (threadIdx.x >> 5);
  if (node >= Nn) return;
  int lane = threadIdx.x & 31;
  const __bf16* xr = xcb + (size_t)node * 512 + lane * 16;
  float xv[16];
  v8bf a0 = *(const v8bf*)xr;
  v8bf a1 = *(const v8bf*)(xr + 8);
#pragma unroll
  for (int j = 0; j < 8; ++j) { xv[j] = (float)a0[j]; xv[8 + j] = (float)a1[j]; }

  float best = -3.4e38f; int bidx = 0;
#pragma unroll
  for (int c = 0; c < 8; ++c) {
    const float* wp = Wp + c * 512 + lane * 16;
    float p = 0.0f;
#pragma unroll
    for (int j = 0; j < 16; ++j) p = fmaf(xv[j], wp[j], p);
#pragma unroll
    for (int off = 16; off > 0; off >>= 1) p += __shfl_xor(p, off);
    if (p > best) { best = p; bidx = c; }  // strict > keeps first-max (argmax semantics)
  }
  const float* web = We + (size_t)bidx * 40 * 512 + lane * 16;
  for (int k = 0; k < 40; ++k) {
    const float* we = web + (size_t)k * 512;
    float p = 0.0f;
#pragma unroll
    for (int j = 0; j < 16; ++j) p = fmaf(xv[j], we[j], p);
#pragma unroll
    for (int off = 16; off > 0; off >>= 1) p += __shfl_xor(p, off);
    if (lane == 0) out[(size_t)node * 40 + k] = p;
  }
}

// ------------------------------------------------------------------- launch
extern "C" void kernel_launch(void* const* d_in, const int* in_sizes, int n_in,
                              void* d_out, int out_size, void* d_ws, size_t ws_size,
                              hipStream_t stream) {
  const float*     x  = (const float*)d_in[0];
  const long long* ei = (const long long*)d_in[1];
  const float*     W1 = (const float*)d_in[2];
  const float*     b1 = (const float*)d_in[3];
  const float*     W2 = (const float*)d_in[4];
  const float*     b2 = (const float*)d_in[5];
  const float*     Wp = (const float*)d_in[6];
  const float*     We = (const float*)d_in[7];

  const int Nn = in_sizes[0] / 512;  // 50000
  const int Ee = in_sizes[1] / 2;    // 1600000

  float* out = (float*)d_out;               // [N,40]
  float* xc  = out + (size_t)Nn * 40;       // [N,512]

  // ---- workspace layout (phases alias; ~212 MB peak) ----
  char* ws = (char*)d_ws;
  unsigned* deg  = (unsigned*)(ws + 0);
  float*    dinv = (float*)  (ws + (1ull  << 20));
  __bf16*   W1t  = (__bf16*) (ws + (2ull  << 20));   // [512][512]
  __bf16*   W2t  = (__bf16*) (ws + (3ull  << 20));   // [256][512]
  __bf16*   Bx   = (__bf16*) (ws + (4ull  << 20));   // x_bf16 -> h1_bf16 -> xc_bf16 [N*512]
  __bf16*   Dh   = (__bf16*) (ws + (56ull << 20));   // H1 bf16 -> H2 bf16
  __bf16*   h2b  = (__bf16*) (ws + (82ull << 20));   // [N*256]
  float*    aggE = (float*)  (ws + (108ull<< 20));   // agg1 [N*512] -> agg2 [N*256]
  float*    nsum = (float*)  (ws + (160ull<< 20));   // [N*256]

  const size_t n512 = (size_t)Nn * 512, n256 = (size_t)Nn * 256;
  const int eb = (Ee + 255) / 256;      // 1 thread/edge
  const int ew = (Ee + 7) / 8;          // 1 wave/edge

  // degrees + norms
  k_zero_u32<<<256, 256, 0, stream>>>(deg, (size_t)Nn);
  k_deg<<<eb, 256, 0, stream>>>(ei, Ee, deg);
  k_dinv<<<(Nn + 255) / 256, 256, 0, stream>>>(deg, dinv, Nn);

  // bf16 operands
  k_f32_to_bf16<<<4096, 256, 0, stream>>>(x, Bx, n512);
  k_tcvt<<<(512 * 512 + 255) / 256, 256, 0, stream>>>(W1, W1t, 512, 512);
  k_tcvt<<<(512 * 256 + 255) / 256, 256, 0, stream>>>(W2, W2t, 512, 256);

  // ---- layer 1: H1 = x @ W1 (WMMA), GCN aggregate, relu ----
  k_zero_f32<<<4096, 256, 0, stream>>>(aggE, n512);
  k_gemm_bf16<<<dim3(2, Nn / 16), 128, 0, stream>>>(Bx, W1t, Dh, Nn, 512, 512);
  k_edge<512, false><<<ew, 256, 0, stream>>>(ei, Ee, Dh, aggE, dinv);
  k_fin1<<<4096, 256, 0, stream>>>(aggE, Dh, dinv, b1, Bx, n512);

  // ---- layer 2: H2 = h1 @ W2 (WMMA), GCN aggregate ----
  k_zero_f32<<<4096, 256, 0, stream>>>(aggE, n256);
  k_gemm_bf16<<<dim3(1, Nn / 16), 128, 0, stream>>>(Bx, W2t, Dh, Nn, 256, 512);
  k_edge<256, false><<<ew, 256, 0, stream>>>(ei, Ee, Dh, aggE, dinv);
  k_fin2<<<4096, 256, 0, stream>>>(aggE, Dh, dinv, b2, xc, h2b, Bx, n256);

  // ---- neighbor mean over h2 ----
  k_zero_f32<<<4096, 256, 0, stream>>>(nsum, n256);
  k_edge<256, true><<<ew, 256, 0, stream>>>(ei, Ee, h2b, nsum, dinv);
  k_fin3<<<4096, 256, 0, stream>>>(nsum, deg, xc, Bx, n256);

  // ---- router + selected expert ----
  k_expert<<<(Nn + 7) / 8, 256, 0, stream>>>(Bx, Wp, We, out, Nn);
}
